// MultiHeadAttention_5772436046377
// MI455X (gfx1250) — matrix operand order repaired and checked by reading, not compiled, not run on hardware
//
#include <hip/hip_runtime.h>
#include <hip/hip_bf16.h>

typedef __attribute__((ext_vector_type(16))) _Float16 v16h;
typedef __attribute__((ext_vector_type(8)))  float    v8f;

#define EMB 768
#define NH  12
#define HD  64
#define BB  8
#define SS  1024

// ---- WMMA fragment index maps (CDNA5 ISA 7.12.2, 16-bit, wave32) ----
// A 16x32 f16: lane L holds row M = L&15.
// lanes 0-15: halves 0-7 -> K=0..7,  halves 8-15 -> K=16..23
// lanes 16-31: halves 0-7 -> K=8..15, halves 8-15 -> K=24..31
__device__ __forceinline__ int a_k(int lane, int h) {
  int base = (lane & 16) ? 8 : 0;
  return (h < 8) ? (h + base) : (h + 8 + base);
}
// B 32x16 f16: lane L holds col N = L&15; same K map as A (a_k)
__device__ __forceinline__ int b_k(int lane, int h) {
  return a_k(lane, h);
}
// C/D f32 16x16: lane L holds N = L&15; VGPR r holds M = r + (L>=16 ? 8 : 0)

__device__ __forceinline__ v16h load_a_mk(const _Float16* p0, int m0, int stride,
                                          int k0, int lane) {
  v16h a;
  const _Float16* p = p0 + (size_t)(m0 + (lane & 15)) * stride + k0;
#pragma unroll
  for (int h = 0; h < 16; ++h) a[h] = p[a_k(lane, h)];
  return a;
}
// B operand from an [n][k] (n-major) array
__device__ __forceinline__ v16h load_b_nk(const _Float16* p0, int n0, int stride,
                                          int k0, int lane) {
  v16h b;
  const _Float16* p = p0 + (size_t)(n0 + (lane & 15)) * stride + k0;
#pragma unroll
  for (int h = 0; h < 16; ++h) b[h] = p[b_k(lane, h)];
  return b;
}
// B operand from a [k][n] (k-major) array
__device__ __forceinline__ v16h load_b_kn(const _Float16* p0, int k0, int stride,
                                          int n0, int lane) {
  v16h b;
  const _Float16* p = p0 + n0 + (lane & 15);
#pragma unroll
  for (int h = 0; h < 16; ++h) b[h] = p[(size_t)(k0 + b_k(lane, h)) * stride];
  return b;
}

__device__ __forceinline__ v8f wmma_f16(v16h a, v16h b, v8f c) {
  return __builtin_amdgcn_wmma_f32_16x16x32_f16(false, a, false, b, (short)0, c,
                                                false, false);
}

// Async 64x64 f16 tile load: global (row stride 64 halves) -> LDS (row stride
// 72 halves), 128b per op, tracked by ASYNCcnt. 4 ops/thread at 128 threads.
__device__ __forceinline__ void async_tile_load(uint32_t lds_base,
                                                const _Float16* gbase,
                                                int tid) {
  for (int i = tid; i < 512; i += 128) {
    int r = i >> 3, c = i & 7;
    uint32_t dst = lds_base + (uint32_t)(r * 9 + c) * 16u;  // 72h=144B=9*16B
    uint32_t src = (uint32_t)(r * 8 + c) * 16u;             // 64h=128B=8*16B
    asm volatile("global_load_async_to_lds_b128 %0, %1, %2"
                 :: "v"(dst), "v"(src), "s"(gbase)
                 : "memory");
  }
}

// ================= Kernel 1: per-head Q/K/V projection (f32 -> f16) ==========
// grid = B*NH*16 blocks, 128 threads. Each block: one (b,h), 64 S-rows,
// three 64x64 GEMMs (y = x @ W^T). Q is pre-scaled by 1/sqrt(d).
__global__ __launch_bounds__(128) void proj_qkv(
    const float* __restrict__ q, const float* __restrict__ k,
    const float* __restrict__ v, const float* __restrict__ Wq,
    const float* __restrict__ Wk, const float* __restrict__ Wv,
    _Float16* __restrict__ Qp, _Float16* __restrict__ Kp,
    _Float16* __restrict__ Vp) {
  __shared__ _Float16 Xl[64 * 72];
  __shared__ _Float16 Wl[64 * 72];
  const int tid = threadIdx.x, lane = tid & 31, wave = tid >> 5;
  const int blk = blockIdx.x;
  const int st = blk & 15;
  const int h  = (blk >> 4) % NH;
  const int b  = blk / (16 * NH);
  const int s0 = st * 64;

  const float* Xsrc[3] = {q, k, v};
  const float* Wsrc[3] = {Wq + h * HD * HD, Wk + h * HD * HD, Wv + h * HD * HD};
  _Float16* Out[3] = {Qp, Kp, Vp};

  for (int p = 0; p < 3; ++p) {
    __syncthreads();
    // Stage X tile [64 rows x 64 d] (head slice of [B,S,E]) as f16
    const float* xb = Xsrc[p] + ((size_t)(b * SS + s0)) * EMB + h * HD;
    for (int i = tid; i < 1024; i += 128) {
      int r = i >> 4, c4 = (i & 15) * 4;
      float4 f = *(const float4*)(xb + (size_t)r * EMB + c4);
      Xl[r * 72 + c4 + 0] = (_Float16)f.x;
      Xl[r * 72 + c4 + 1] = (_Float16)f.y;
      Xl[r * 72 + c4 + 2] = (_Float16)f.z;
      Xl[r * 72 + c4 + 3] = (_Float16)f.w;
    }
    // Stage W [out n][in k] as f16
    const float* wb = Wsrc[p];
    for (int i = tid; i < 1024; i += 128) {
      int r = i >> 4, c4 = (i & 15) * 4;
      float4 f = *(const float4*)(wb + r * HD + c4);
      Wl[r * 72 + c4 + 0] = (_Float16)f.x;
      Wl[r * 72 + c4 + 1] = (_Float16)f.y;
      Wl[r * 72 + c4 + 2] = (_Float16)f.z;
      Wl[r * 72 + c4 + 3] = (_Float16)f.w;
    }
    __syncthreads();

    const float scl = (p == 0) ? 0.125f : 1.0f;  // fold 1/sqrt(64) into Q
    _Float16* outp = Out[p] + ((size_t)(b * NH + h) * SS + s0) * HD;
    const int mtile = wave * 16;
    const int nloc = lane & 15;
    const int mhi = (lane & 16) ? 8 : 0;
#pragma unroll
    for (int nt = 0; nt < 4; ++nt) {
      v8f c = {};
#pragma unroll
      for (int kc = 0; kc < 2; ++kc) {
        v16h a = load_a_mk(Xl, mtile, 72, kc * 32, lane);
        v16h bfr = load_b_nk(Wl, nt * 16, 72, kc * 32, lane);
        c = wmma_f16(a, bfr, c);
      }
#pragma unroll
      for (int r = 0; r < 8; ++r)
        outp[(size_t)(mtile + mhi + r) * HD + nt * 16 + nloc] =
            (_Float16)(c[r] * scl);
    }
  }
}

// ================= Kernel 2: flash attention per (b,h,64 q-rows) =============
// grid = B*NH*16, 128 threads = 4 waves; wave owns 16 q-rows x all 64 d.
// K/V tiles double-buffered in LDS, filled by global_load_async_to_lds_b128
// (ASYNCcnt) so tile t+1 streams from HBM while tile t is computed.
__global__ __launch_bounds__(128) void attn_fused(
    const _Float16* __restrict__ Qp, const _Float16* __restrict__ Kp,
    const _Float16* __restrict__ Vp, _Float16* __restrict__ Xo) {
  __shared__ __align__(16) _Float16 KtBuf[2][64 * 72];
  __shared__ __align__(16) _Float16 VtBuf[2][64 * 72];
  __shared__ __align__(16) _Float16 Pl[4 * 16 * 72];  // per-wave 16x64 P tile
  const int tid = threadIdx.x, lane = tid & 31, wave = tid >> 5;
  const int blk = blockIdx.x;
  const int qt = blk & 15;
  const int h  = (blk >> 4) % NH;
  const int b  = blk / (16 * NH);
  const size_t bh = (size_t)(b * NH + h) * SS;
  const int q0 = qt * 64 + wave * 16;

  // Q A-fragments (16 q-rows x 64 d, two K=32 chunks), loaded once
  const _Float16* qbase = Qp + (bh + q0) * HD;
  v16h aQ0 = load_a_mk(qbase, 0, HD, 0, lane);
  v16h aQ1 = load_a_mk(qbase, 0, HD, 32, lane);

  v8f acc[4] = {};
  float mx[8], ls[8];
#pragma unroll
  for (int r = 0; r < 8; ++r) { mx[r] = -1e30f; ls[r] = 0.0f; }
  _Float16* Pw = Pl + wave * 16 * 72;
  const int nloc = lane & 15;
  const int mhi = (lane & 16) ? 8 : 0;

  const uint32_t ldsK[2] = {(uint32_t)(uintptr_t)&KtBuf[0][0],
                            (uint32_t)(uintptr_t)&KtBuf[1][0]};
  const uint32_t ldsV[2] = {(uint32_t)(uintptr_t)&VtBuf[0][0],
                            (uint32_t)(uintptr_t)&VtBuf[1][0]};

  // prologue: kick off tile 0 into buffer 0
  async_tile_load(ldsK[0], Kp + bh * HD, tid);
  async_tile_load(ldsV[0], Vp + bh * HD, tid);

  for (int t = 0; t < 16; ++t) {
    const int cur = t & 1;
    // wait for this wave's async fills of buffer `cur`, then make visible
    asm volatile("s_wait_asynccnt 0" ::: "memory");
    __syncthreads();
    // buffer cur^1 was last read in iteration t-1 (all waves past barrier):
    // safe to start overwriting it with tile t+1 while we compute tile t.
    if (t < 15) {
      async_tile_load(ldsK[cur ^ 1], Kp + (bh + (t + 1) * 64) * HD, tid);
      async_tile_load(ldsV[cur ^ 1], Vp + (bh + (t + 1) * 64) * HD, tid);
    }
    const _Float16* Kt = &KtBuf[cur][0];
    const _Float16* Vt = &VtBuf[cur][0];

    // scores S[16q x 64kv] = Q Kt^T (Q pre-scaled by 1/sqrt(d))
    v8f s[4];
#pragma unroll
    for (int kv = 0; kv < 4; ++kv) {
      v8f c0 = {};
      c0 = wmma_f16(aQ0, load_b_nk(Kt, kv * 16, 72, 0, lane), c0);
      c0 = wmma_f16(aQ1, load_b_nk(Kt, kv * 16, 72, 32, lane), c0);
      s[kv] = c0;
    }

    // online softmax: row m lives across 16 lanes (fixed VGPR r, half-wave)
    float pm[8];
#pragma unroll
    for (int r = 0; r < 8; ++r)
      pm[r] = fmaxf(fmaxf(s[0][r], s[1][r]), fmaxf(s[2][r], s[3][r]));
#pragma unroll
    for (int off = 1; off < 16; off <<= 1)
#pragma unroll
      for (int r = 0; r < 8; ++r)
        pm[r] = fmaxf(pm[r], __shfl_xor(pm[r], off, 32));

    float alpha[8];
#pragma unroll
    for (int r = 0; r < 8; ++r) {
      float mn = fmaxf(mx[r], pm[r]);
      alpha[r] = __expf(mx[r] - mn);
      mx[r] = mn;
    }
    float rs[8];
#pragma unroll
    for (int r = 0; r < 8; ++r) rs[r] = 0.0f;
#pragma unroll
    for (int kv = 0; kv < 4; ++kv) {
#pragma unroll
      for (int r = 0; r < 8; ++r) {
        float p = __expf(s[kv][r] - mx[r]);
        rs[r] += p;
        // transpose C-layout -> A-layout via wave-private LDS
        Pw[(size_t)(mhi + r) * 72 + kv * 16 + nloc] = (_Float16)p;
      }
    }
#pragma unroll
    for (int off = 1; off < 16; off <<= 1)
#pragma unroll
      for (int r = 0; r < 8; ++r) rs[r] += __shfl_xor(rs[r], off, 32);
#pragma unroll
    for (int r = 0; r < 8; ++r) ls[r] = ls[r] * alpha[r] + rs[r];
#pragma unroll
    for (int d = 0; d < 4; ++d)
#pragma unroll
      for (int r = 0; r < 8; ++r) acc[d][r] *= alpha[r];

    // acc += P @ V   (LDS ops from the same wave stay in order)
    v16h aP0 = load_a_mk(Pw, 0, 72, 0, lane);
    v16h aP1 = load_a_mk(Pw, 0, 72, 32, lane);
#pragma unroll
    for (int d = 0; d < 4; ++d) {
      acc[d] = wmma_f16(aP0, load_b_kn(Vt, 0, 72, d * 16, lane), acc[d]);
      acc[d] = wmma_f16(aP1, load_b_kn(Vt, 32, 72, d * 16, lane), acc[d]);
    }
  }

  // normalize and write merged-head f16 activations [B,S,E]
  float inv[8];
#pragma unroll
  for (int r = 0; r < 8; ++r) inv[r] = 1.0f / ls[r];
#pragma unroll
  for (int d = 0; d < 4; ++d)
#pragma unroll
    for (int r = 0; r < 8; ++r)
      Xo[((size_t)b * SS + q0 + mhi + r) * EMB + h * HD + d * 16 + nloc] =
          (_Float16)(acc[d][r] * inv[r]);
}

// ================= Kernel 3: output projection + bias (f32 out) ==============
// out[8192,768] = X[8192,768] @ Wo^T + bo. grid = 128*12, 64x64 tile per block.
__global__ __launch_bounds__(128) void out_proj(
    const _Float16* __restrict__ X, const float* __restrict__ Wo,
    const float* __restrict__ bo, float* __restrict__ out) {
  __shared__ __align__(16) _Float16 Xl[64 * 48];  // 64 rows x 32 k
  __shared__ __align__(16) _Float16 Wl[64 * 48];  // 64 n    x 32 k
  const int tid = threadIdx.x, lane = tid & 31, wave = tid >> 5;
  const int nt = blockIdx.x % (EMB / 64);
  const int rt = blockIdx.x / (EMB / 64);
  const size_t row0 = (size_t)rt * 64;
  const int n0 = nt * 64;

  v8f acc[4] = {};
  for (int kc = 0; kc < EMB / 32; ++kc) {
    const int k0 = kc * 32;
    __syncthreads();
    // X tile: 64 rows x 32 halves (4 uint4 per row)
    for (int i = tid; i < 256; i += 128) {
      int r = i >> 2, c = i & 3;
      *(uint4*)(Xl + r * 48 + c * 8) =
          *(const uint4*)(X + (row0 + r) * (size_t)EMB + k0 + c * 8);
    }
    // W tile: Wo[n0+n][k0..k0+31] f32 -> f16
    for (int i = tid; i < 512; i += 128) {
      int r = i >> 3, c4 = (i & 7) * 4;
      float4 f = *(const float4*)(Wo + (size_t)(n0 + r) * EMB + k0 + c4);
      Wl[r * 48 + c4 + 0] = (_Float16)f.x;
      Wl[r * 48 + c4 + 1] = (_Float16)f.y;
      Wl[r * 48 + c4 + 2] = (_Float16)f.z;
      Wl[r * 48 + c4 + 3] = (_Float16)f.w;
    }
    __syncthreads();
    v16h a = load_a_mk(Xl, wave * 16, 48, 0, lane);
#pragma unroll
    for (int n = 0; n < 4; ++n)
      acc[n] = wmma_f16(a, load_b_nk(Wl, n * 16, 48, 0, lane), acc[n]);
  }

  const int nloc = lane & 15;
  const int mhi = (lane & 16) ? 8 : 0;
#pragma unroll
  for (int n = 0; n < 4; ++n) {
    float bias = bo[n0 + n * 16 + nloc];
#pragma unroll
    for (int r = 0; r < 8; ++r)
      out[(row0 + wave * 16 + mhi + r) * (size_t)EMB + n0 + n * 16 + nloc] =
          acc[n][r] + bias;
  }
}

extern "C" void kernel_launch(void* const* d_in, const int* in_sizes, int n_in,
                              void* d_out, int out_size, void* d_ws,
                              size_t ws_size, hipStream_t stream) {
  (void)in_sizes; (void)n_in; (void)out_size; (void)ws_size;
  const float* q  = (const float*)d_in[0];
  const float* k  = (const float*)d_in[1];
  const float* v  = (const float*)d_in[2];
  const float* Wq = (const float*)d_in[3];
  const float* Wk = (const float*)d_in[4];
  const float* Wv = (const float*)d_in[5];
  const float* Wo = (const float*)d_in[6];
  const float* bo = (const float*)d_in[7];

  const size_t nqkv = (size_t)BB * NH * SS * HD;  // 6.29M halves each
  _Float16* Qp = (_Float16*)d_ws;
  _Float16* Kp = Qp + nqkv;
  _Float16* Vp = Kp + nqkv;
  _Float16* Xo = Vp + nqkv;  // [B,S,E] f16 merged-head activations

  proj_qkv<<<BB * NH * 16, 128, 0, stream>>>(q, k, v, Wq, Wk, Wv, Qp, Kp, Vp);
  attn_fused<<<BB * NH * 16, 128, 0, stream>>>(Qp, Kp, Vp, Xo);
  out_proj<<<(SS * BB / 64) * (EMB / 64), 128, 0, stream>>>(Xo, Wo, bo,
                                                            (float*)d_out);
}
